// LSTM_45028437131359
// MI455X (gfx1250) — matrix operand order repaired and checked by reading, not compiled, not run on hardware
//
#include <hip/hip_runtime.h>
#include <hip/hip_bf16.h>

typedef __attribute__((ext_vector_type(16))) __bf16 v16bf;
typedef __attribute__((ext_vector_type(8)))  __bf16 v8bf;
typedef __attribute__((ext_vector_type(8)))  float  v8f;

#define B_  512
#define T_  512
#define H_  256
#define O_  2
#define NT  512     // threads per block (16 waves)

__device__ __forceinline__ unsigned short f2bf(float f) {
  unsigned u = __float_as_uint(f);
  u += 0x7FFFu + ((u >> 16) & 1u);          // round-to-nearest-even
  return (unsigned short)(u >> 16);
}
__device__ __forceinline__ float bf2f(unsigned short h) {
  return __uint_as_float(((unsigned)h) << 16);
}
__device__ __forceinline__ float sigm(float x)      { return 1.f / (1.f + __expf(-x)); }
__device__ __forceinline__ float tanh_fast(float x) { return 2.f / (1.f + __expf(-2.f * x)) - 1.f; }

// A-matrix fragment, 16x32 bf16, row-major source with ld=H_.
// ISA layout: lanes 0-15 row m, K {0..7,16..23}; lanes 16-31 row m, K {8..15,24..31}.
__device__ __forceinline__ v16bf load_a16x32(const unsigned short* base, int kb, int lane) {
  int row = lane & 15;
  int hh  = lane >> 4;
  const unsigned short* p = base + row * H_ + kb + hh * 8;
  v8bf lo = *(const v8bf*)(p);        // K = kb + hh*8 .. +7
  v8bf hi = *(const v8bf*)(p + 16);   // K = kb + hh*8 + 16 .. +23
  v16bf f;
#pragma unroll
  for (int i = 0; i < 8; ++i) { f[i] = lo[i]; f[8 + i] = hi[i]; }
  return f;
}

// B-matrix fragment, 32x16 bf16. B[k][n] = W[n][k], W row-major (H_,H_).
// Layout: lanes 0-15 col n hold K {kb..kb+7, kb+16..kb+23}; lanes 16-31 hold K {kb+8..kb+15, kb+24..kb+31}.
__device__ __forceinline__ v16bf load_b32x16(const unsigned short* Wg, int rowbase, int kb, int lane) {
  int n  = lane & 15;
  int kh = lane >> 4;
  const unsigned short* p = Wg + (rowbase + n) * H_ + kb + kh * 8;
  v8bf lo = *(const v8bf*)(p);
  v8bf hi = *(const v8bf*)(p + 16);
  v16bf f;
#pragma unroll
  for (int i = 0; i < 8; ++i) { f[i] = lo[i]; f[8 + i] = hi[i]; }
  return f;
}

__device__ __forceinline__ v8f wmma_bf16(v16bf a, v16bf b, v8f c) {
  return __builtin_amdgcn_wmma_f32_16x16x32_bf16(false, a, false, b, (short)0, c, false, false);
}

// ---------------- weight f32 -> bf16 packing ----------------
struct ConvArgs { const float* src[12]; unsigned short* dst; };

__global__ void convert_weights(ConvArgs a) {
  int idx = blockIdx.x * blockDim.x + threadIdx.x;
  if (idx >= 12 * 65536) return;
  int arr = idx >> 16;
  int off = idx & 65535;
  a.dst[idx] = f2bf(a.src[arr][off]);
}

// ---------------- persistent LSTM kernel ----------------
struct LstmArgs {
  const float* x;               // [B_,T_] f32
  const unsigned short* w;      // bf16: [Wh0 i,f,g,o][Wi1 i,f,g,o][Wh1 i,f,g,o], each 256*256
  const float* wi0[4];          // layer-0 input weights (H_,1), gates i,f,g,o
  const float* b0in[4];         // layer-0 input biases
  const float* b0h[4];          // layer-0 hidden biases
  const float* b1in[4];         // layer-1 input biases
  const float* b1h[4];          // layer-1 hidden biases
  const float* wfc;             // (O_,H_)
  const float* bfc;             // (O_,)
  float* out;                   // (B_,O_)
};

__global__ __launch_bounds__(NT, 1) void lstm_persistent(LstmArgs A) {
  __shared__ __align__(16) unsigned short h0[2][16 * H_];
  __shared__ __align__(16) unsigned short h1[2][16 * H_];
  __shared__ float xbuf[16];

  const int tid  = threadIdx.x;
  const int lane = tid & 31;
  const int wave = tid >> 5;
  const int b0   = blockIdx.x * 16;
  const int jrow = wave * 16;        // this wave's hidden-unit tile
  const int n    = lane & 15;
  const int hh   = lane >> 4;
  const int j    = jrow + n;

  // init: zero starting h buffers, prefetch x_0
  for (int i = tid; i < 16 * H_; i += NT) { h0[0][i] = 0; h1[0][i] = 0; }
  if (wave == 0 && lane < 16) xbuf[lane] = A.x[(b0 + lane) * T_ + 0];

  // hoisted per-lane (per-j) scalars
  float wi0g[4], bia0[4], bia1[4];
#pragma unroll
  for (int g = 0; g < 4; ++g) {
    wi0g[g] = A.wi0[g][j];
    bia0[g] = A.b0in[g][j] + A.b0h[g][j];
    bia1[g] = A.b1in[g][j] + A.b1h[g][j];
  }
  const unsigned short* Wh0[4];
  const unsigned short* Wi1[4];
  const unsigned short* Wh1[4];
#pragma unroll
  for (int g = 0; g < 4; ++g) {
    Wh0[g] = A.w + (0 + g) * 65536;
    Wi1[g] = A.w + (4 + g) * 65536;
    Wh1[g] = A.w + (8 + g) * 65536;
  }

  float c0r[8], c1r[8];
#pragma unroll
  for (int r = 0; r < 8; ++r) { c0r[r] = 0.f; c1r[r] = 0.f; }

  __syncthreads();

  int p = 0;
  for (int t = 0; t < T_; ++t) {
    // ---------------- layer 0: preact = h0_prev @ Wh0^T (+ x-term, bias) ----------------
    v8f ai = {}, af = {}, ag = {}, ao = {};
#pragma unroll
    for (int kb = 0; kb < H_; kb += 32) {
      v16bf a = load_a16x32(h0[p], kb, lane);
      ai = wmma_bf16(a, load_b32x16(Wh0[0], jrow, kb, lane), ai);
      af = wmma_bf16(a, load_b32x16(Wh0[1], jrow, kb, lane), af);
      ag = wmma_bf16(a, load_b32x16(Wh0[2], jrow, kb, lane), ag);
      ao = wmma_bf16(a, load_b32x16(Wh0[3], jrow, kb, lane), ao);
    }
    {
      const float* xb = xbuf + hh * 8;
#pragma unroll
      for (int r = 0; r < 8; ++r) {
        float xt = xb[r];
        float iv = sigm(ai[r] + bia0[0] + xt * wi0g[0]);
        float fv = sigm(af[r] + bia0[1] + xt * wi0g[1]);
        float gv = tanh_fast(ag[r] + bia0[2] + xt * wi0g[2]);
        float ov = sigm(ao[r] + bia0[3] + xt * wi0g[3]);
        float c  = fv * c0r[r] + iv * gv;
        c0r[r]   = c;
        float hv = ov * tanh_fast(c);
        h0[p ^ 1][(r + hh * 8) * H_ + j] = f2bf(hv);
      }
    }
    __syncthreads();   // h0_new visible; xbuf free for prefetch

    // ---------------- layer 1: preact = h0_new @ Wi1^T + h1_prev @ Wh1^T + bias ----------------
    ai = {}; af = {}; ag = {}; ao = {};
#pragma unroll
    for (int kb = 0; kb < H_; kb += 32) {
      v16bf a = load_a16x32(h0[p ^ 1], kb, lane);
      ai = wmma_bf16(a, load_b32x16(Wi1[0], jrow, kb, lane), ai);
      af = wmma_bf16(a, load_b32x16(Wi1[1], jrow, kb, lane), af);
      ag = wmma_bf16(a, load_b32x16(Wi1[2], jrow, kb, lane), ag);
      ao = wmma_bf16(a, load_b32x16(Wi1[3], jrow, kb, lane), ao);
    }
#pragma unroll
    for (int kb = 0; kb < H_; kb += 32) {
      v16bf a = load_a16x32(h1[p], kb, lane);
      ai = wmma_bf16(a, load_b32x16(Wh1[0], jrow, kb, lane), ai);
      af = wmma_bf16(a, load_b32x16(Wh1[1], jrow, kb, lane), af);
      ag = wmma_bf16(a, load_b32x16(Wh1[2], jrow, kb, lane), ag);
      ao = wmma_bf16(a, load_b32x16(Wh1[3], jrow, kb, lane), ao);
    }
    if (wave == 0 && lane < 16 && t + 1 < T_)
      xbuf[lane] = A.x[(b0 + lane) * T_ + (t + 1)];
#pragma unroll
    for (int r = 0; r < 8; ++r) {
      float iv = sigm(ai[r] + bia1[0]);
      float fv = sigm(af[r] + bia1[1]);
      float gv = tanh_fast(ag[r] + bia1[2]);
      float ov = sigm(ao[r] + bia1[3]);
      float c  = fv * c1r[r] + iv * gv;
      c1r[r]   = c;
      float hv = ov * tanh_fast(c);
      h1[p ^ 1][(r + hh * 8) * H_ + j] = f2bf(hv);
    }
    __syncthreads();   // h1_new visible; old buffers free
    p ^= 1;
  }

  // ---------------- final FC: out[b, o] = h1_last[b] . wfc[o] + bfc[o] ----------------
  if (wave == 0) {
    int m = lane >> 1;
    int o = lane & 1;
    const unsigned short* hr = h1[p] + m * H_;
    const float* wr = A.wfc + o * H_;
    float s = A.bfc[o];
    for (int k = 0; k < H_; ++k) s += bf2f(hr[k]) * wr[k];
    A.out[(b0 + m) * O_ + o] = s;
  }
}

extern "C" void kernel_launch(void* const* d_in, const int* in_sizes, int n_in,
                              void* d_out, int out_size, void* d_ws, size_t ws_size,
                              hipStream_t stream) {
  (void)in_sizes; (void)n_in; (void)out_size; (void)ws_size;
  // setup_inputs() insertion order:
  // 0:x  | W_ii: 1,2 (l0 W,b) 3,4 (l1 W,b) | W_if: 5..8 | W_ig: 9..12 | W_io: 13..16
  // W_hi: 17..20 | W_hf: 21..24 | W_hg: 25..28 | W_ho: 29..32 | fc: 33 (W), 34 (b)
  static const int wh0_i[4]  = {17, 21, 25, 29};
  static const int wi1_i[4]  = { 3,  7, 11, 15};
  static const int wh1_i[4]  = {19, 23, 27, 31};
  static const int wi0_i[4]  = { 1,  5,  9, 13};
  static const int b0in_i[4] = { 2,  6, 10, 14};
  static const int b0h_i[4]  = {18, 22, 26, 30};
  static const int b1in_i[4] = { 4,  8, 12, 16};
  static const int b1h_i[4]  = {20, 24, 28, 32};

  ConvArgs ca;
  for (int g = 0; g < 4; ++g) {
    ca.src[g]     = (const float*)d_in[wh0_i[g]];
    ca.src[4 + g] = (const float*)d_in[wi1_i[g]];
    ca.src[8 + g] = (const float*)d_in[wh1_i[g]];
  }
  ca.dst = (unsigned short*)d_ws;
  convert_weights<<<dim3(3072), dim3(256), 0, stream>>>(ca);

  LstmArgs la;
  la.x = (const float*)d_in[0];
  la.w = (const unsigned short*)d_ws;
  for (int g = 0; g < 4; ++g) {
    la.wi0[g]  = (const float*)d_in[wi0_i[g]];
    la.b0in[g] = (const float*)d_in[b0in_i[g]];
    la.b0h[g]  = (const float*)d_in[b0h_i[g]];
    la.b1in[g] = (const float*)d_in[b1in_i[g]];
    la.b1h[g]  = (const float*)d_in[b1h_i[g]];
  }
  la.wfc = (const float*)d_in[33];
  la.bfc = (const float*)d_in[34];
  la.out = (float*)d_out;
  lstm_persistent<<<dim3(B_ / 16), dim3(NT), 0, stream>>>(la);
}
